// MixerModel_1846835938125
// MI455X (gfx1250) — matrix operand order repaired and checked by reading, compiled, not run on hardware
//
#include <hip/hip_runtime.h>
#include <hip/hip_bf16.h>

// ---------------------------------------------------------------------------
// Types for CDNA5 WMMA (wave32): v16bf A/B operands, v8f accumulator
// ---------------------------------------------------------------------------
typedef __attribute__((ext_vector_type(16))) __bf16 v16bf;
typedef __attribute__((ext_vector_type(8)))  float  v8f;
// TDM descriptor groups (probe-verified arg types for tensor_load_to_lds)
typedef __attribute__((ext_vector_type(4))) unsigned v4u;
typedef __attribute__((ext_vector_type(8))) int      v8i_;
typedef __attribute__((ext_vector_type(4))) int      v4i_;

#if defined(__AMDGCN__) && __has_builtin(__builtin_amdgcn_tensor_load_to_lds)
#define USE_TDM 1
#else
#define USE_TDM 0
#endif

struct alignas(16) Q128 { unsigned u[4]; };

union AF {                 // one WMMA A/B fragment (32 bytes per lane)
    Q128   q[2];
    v16bf  v;
    __bf16 h[16];
};

__device__ __forceinline__ __bf16 f2bf(float f) {
    union { float f; unsigned u; } in; in.f = f;
    unsigned u = in.u;
    unsigned r = u + 0x7FFFu + ((u >> 16) & 1u);   // round-to-nearest-even
    union { unsigned short s; __bf16 b; } out;
    out.s = (unsigned short)(r >> 16);
    return out.b;
}

__device__ __forceinline__ float sigmoidf_(float x) {
    return 1.0f / (1.0f + __expf(-x));
}

// Model dims
#define D_MODEL 512
#define D_INNER 1024
#define TLEN    1024
#define BATCH   4
#define MROWS   (BATCH * TLEN)   // 4096

// ---------------------------------------------------------------------------
// zero kernel (workspace init: h-state, barrier counter)
// ---------------------------------------------------------------------------
__global__ void k_zero(float* __restrict__ p, int n) {
    int i = blockIdx.x * blockDim.x + threadIdx.x;
    int stride = gridDim.x * blockDim.x;
    for (; i < n; i += stride) p[i] = 0.0f;
}

// ---------------------------------------------------------------------------
// transposing f32 -> bf16 weight conversion: src (N,K) -> dst (K,N)
// (done once per layer; lets the GEMM stage BOTH operands with the TDM)
// ---------------------------------------------------------------------------
__global__ void k_cvt_t(const float* __restrict__ src, __bf16* __restrict__ dst,
                        int N, int K) {
    int i = blockIdx.x * blockDim.x + threadIdx.x;
    int stride = gridDim.x * blockDim.x;
    const int total = N * K;
    for (; i < total; i += stride) {
        const int k = i / N;
        const int n = i - k * N;
        dst[i] = f2bf(src[(size_t)n * K + k]);   // coalesced writes
    }
}

// ---------------------------------------------------------------------------
// residual add + layernorm. One block per row of 512.
//   first==1 : resid = hin          else resid += hin
//   bf16mode : write bf16 (GEMM A)  else write f32 (final output)
// ---------------------------------------------------------------------------
__global__ __launch_bounds__(256)
void k_resid_ln(const float* __restrict__ hin, float* __restrict__ resid,
                __bf16* __restrict__ obf, float* __restrict__ of32,
                const float* __restrict__ w, const float* __restrict__ b,
                int first, int bf16mode)
{
    __shared__ float s1[256];
    __shared__ float s2[256];
    const int row = blockIdx.x, tid = threadIdx.x;
    const float* hr = hin + (size_t)row * D_MODEL;
    float* rr = resid + (size_t)row * D_MODEL;

    float v0 = hr[tid], v1 = hr[tid + 256];
    if (!first) { v0 += rr[tid]; v1 += rr[tid + 256]; }
    rr[tid] = v0; rr[tid + 256] = v1;

    s1[tid] = v0 + v1;
    s2[tid] = v0 * v0 + v1 * v1;
    __syncthreads();
    for (int s = 128; s > 0; s >>= 1) {
        if (tid < s) { s1[tid] += s1[tid + s]; s2[tid] += s2[tid + s]; }
        __syncthreads();
    }
    const float mean = s1[0] * (1.0f / D_MODEL);
    const float var  = s2[0] * (1.0f / D_MODEL) - mean * mean;
    const float inv  = rsqrtf(var + 1e-5f);

    const float y0 = (v0 - mean) * inv * w[tid]       + b[tid];
    const float y1 = (v1 - mean) * inv * w[tid + 256] + b[tid + 256];
    if (bf16mode) {
        obf[(size_t)row * D_MODEL + tid]       = f2bf(y0);
        obf[(size_t)row * D_MODEL + tid + 256] = f2bf(y1);
    } else {
        of32[(size_t)row * D_MODEL + tid]       = y0;
        of32[(size_t)row * D_MODEL + tid + 256] = y1;
    }
}

// ---------------------------------------------------------------------------
// Tiled WMMA bf16 GEMM:  C(M,N) f32 = A(M,K) bf16 x WT(K,N) bf16 + bias
// BM=BN=128, BK=32, 256 threads = 8 waves in 2(M) x 4(N), each wave 64x32.
// BOTH tiles staged by the Tensor Data Mover (tensor_load_to_lds, TENSORcnt),
// double-buffered: wait tile kt -> barrier -> issue DMAs for kt+1 -> compute.
// D# LDS padding gives conflict-free strides (A: 64B+16B, WT: 256B+16B).
// Inner loop: ds_load_b128 for A, per-element 16-bit column gathers for WT -> v_wmma, DMA overlapped.
// ---------------------------------------------------------------------------
#define BM 128
#define BN 128
#define BK 32
#define ASTR 40      // As row stride (bf16) == 64B data + 16B pad
#define WTSTR 136    // Wt row stride (bf16) == 256B data + 16B pad

#if USE_TDM
typedef __attribute__((address_space(3))) __bf16 bf16_lds;

// One TDM 2D tile load from a row-major (dim1, dim0) bf16 tensor.
// tile = tile1 rows x tile0 elems; LDS pad per D# (ISA 8.3/8.4).
__device__ __forceinline__ void tdm_load_2d(const __bf16* gsrc, unsigned lds_off,
                                            unsigned dim0, unsigned dim1,
                                            unsigned tile0, unsigned tile1,
                                            unsigned pad_int, unsigned pad_amt)
{
    unsigned long long ga = (unsigned long long)(const void*)gsrc;
    v4u g0;
    g0[0] = 1u;                                            // count=1, user mode
    g0[1] = lds_off;                                       // lds_addr (bytes)
    g0[2] = (unsigned)(ga & 0xFFFFFFFFull);                // global_addr[31:0]
    g0[3] = (unsigned)((ga >> 32) & 0x01FFFFFFull)         // global_addr[56:32]
          | (2u << 30);                                    // type=2 ("image")
    v8i_ g1;
    g1[0] = (int)((1u << 16)            // data_size = 1 -> 2 bytes
                | (1u << 20)            // pad_enable
                | (pad_int << 22)       // pad_interval (2^(v+1) DWORDs data)
                | (pad_amt << 25));     // pad_amount  (v+1 DWORDs pad)
    g1[1] = (int)((dim0 & 0xFFFFu) << 16);                 // tensor_dim0 lo16
    g1[2] = (int)(((dim0 >> 16) & 0xFFFFu)                 // tensor_dim0 hi16
                | ((dim1 & 0xFFFFu) << 16));               // tensor_dim1 lo16
    g1[3] = (int)(((dim1 >> 16) & 0xFFFFu)                 // tensor_dim1 hi16
                | (tile0 << 16));                          // tile_dim0
    g1[4] = (int)tile1;                                    // tile_dim1, dim2=0
    g1[5] = (int)dim0;                                     // dim0_stride lo32
    g1[6] = 0;                                             // stride hi / dim1_stride
    g1[7] = 0;
    v4i_ gz = {0, 0, 0, 0};
#if __clang_major__ >= 23
    v8i_ gz8 = {0, 0, 0, 0, 0, 0, 0, 0};
    __builtin_amdgcn_tensor_load_to_lds(g0, g1, gz, gz, gz8, 0);
#else
    __builtin_amdgcn_tensor_load_to_lds(g0, g1, gz, gz, 0);
#endif
}
#endif // USE_TDM

__global__ __launch_bounds__(256)
void k_gemm(const __bf16* __restrict__ A, const __bf16* __restrict__ WT,
            const float* __restrict__ bias, float* __restrict__ C,
            int N, int K)
{
    __shared__ __bf16 As[2][BM * ASTR];     // 2 x 10240 B
    __shared__ __bf16 Wt[2][BK * WTSTR];    // 2 x  8704 B

    const int tid  = threadIdx.x;
    const int lane = tid & 31;
    const int wid  = tid >> 5;
    const int wm   = wid >> 2;           // 0..1
    const int wn   = wid & 3;            // 0..3
    const int m0   = blockIdx.y * BM;
    const int n0   = blockIdx.x * BN;

    v8f acc[4][2];
#pragma unroll
    for (int mi = 0; mi < 4; mi++)
#pragma unroll
        for (int ni = 0; ni < 2; ni++)
            acc[mi][ni] = (v8f){0.f,0.f,0.f,0.f,0.f,0.f,0.f,0.f};

    const int rsel = lane & 15;
    const int kb   = (lane >> 4) * 8;
    const int nK   = K / BK;

#if USE_TDM
    const unsigned ldsAb[2] = {
        (unsigned)(unsigned long long)(bf16_lds*)&As[0][0],
        (unsigned)(unsigned long long)(bf16_lds*)&As[1][0] };
    const unsigned ldsWb[2] = {
        (unsigned)(unsigned long long)(bf16_lds*)&Wt[0][0],
        (unsigned)(unsigned long long)(bf16_lds*)&Wt[1][0] };

    if (wid == 0) {   // prologue DMAs into buffer 0 (EXEC ignored by TDM)
        tdm_load_2d(&A[(size_t)m0 * K], ldsAb[0], (unsigned)K, MROWS, BK, BM, 3u, 3u);
        tdm_load_2d(&WT[n0],            ldsWb[0], (unsigned)N, (unsigned)K, BN, BK, 5u, 3u);
    }
    for (int kt = 0; kt < nK; kt++) {
        const int cb = kt & 1;
        if (wid == 0)
            __builtin_amdgcn_s_wait_tensorcnt(0);   // tile kt landed in LDS
        __syncthreads();
        if (wid == 0 && kt + 1 < nK) {              // overlap next tile's DMA
            const int k1 = (kt + 1) * BK;
            tdm_load_2d(&A[(size_t)m0 * K + k1], ldsAb[cb ^ 1],
                        (unsigned)K, MROWS, BK, BM, 3u, 3u);
            tdm_load_2d(&WT[(size_t)k1 * N + n0], ldsWb[cb ^ 1],
                        (unsigned)N, (unsigned)K, BN, BK, 5u, 3u);
        }
        AF afr[4], wfr[2];
#pragma unroll
        for (int mi = 0; mi < 4; mi++) {
            const int r = wm * 64 + mi * 16 + rsel;
            afr[mi].q[0] = *(const Q128*)&As[cb][r * ASTR + kb];
            afr[mi].q[1] = *(const Q128*)&As[cb][r * ASTR + kb + 16];
        }
#pragma unroll
        for (int ni = 0; ni < 2; ni++) {
            const int nb = wn * 32 + ni * 16;
            for (int e = 0; e < 16; e++)
                wfr[ni].h[e] = Wt[cb][(kb + (e < 8 ? e : e + 8)) * WTSTR + nb + rsel];
        }
#pragma unroll
        for (int mi = 0; mi < 4; mi++)
#pragma unroll
            for (int ni = 0; ni < 2; ni++)
                acc[mi][ni] = __builtin_amdgcn_wmma_f32_16x16x32_bf16(
                    false, afr[mi].v, false, wfr[ni].v,
                    (short)0, acc[mi][ni], false, false);
    }
#else
    // fallback: manual single-buffer staging (both operands contiguous copies)
    for (int kt = 0; kt < nK; kt++) {
        const int k0 = kt * BK;
#pragma unroll
        for (int i = 0; i < 2; i++) {
            const int j  = tid + i * 256;              // 0..511
            const int ar = j >> 2, ac = (j & 3) * 8;   // A: 128 rows x 4 quads
            *(Q128*)&As[0][ar * ASTR + ac] =
                *(const Q128*)&A[(size_t)(m0 + ar) * K + k0 + ac];
            const int wr = j >> 4, wc = (j & 15) * 8;  // WT: 32 rows x 16 quads
            *(Q128*)&Wt[0][wr * WTSTR + wc] =
                *(const Q128*)&WT[(size_t)(k0 + wr) * N + n0 + wc];
            if (i == 0 && kt + 1 < nK)
                __builtin_prefetch((const void*)&WT[(size_t)(k0 + BK + wr) * N + n0], 0, 0);
        }
        __syncthreads();
        AF afr[4], wfr[2];
#pragma unroll
        for (int mi = 0; mi < 4; mi++) {
            const int r = wm * 64 + mi * 16 + rsel;
            afr[mi].q[0] = *(const Q128*)&As[0][r * ASTR + kb];
            afr[mi].q[1] = *(const Q128*)&As[0][r * ASTR + kb + 16];
        }
#pragma unroll
        for (int ni = 0; ni < 2; ni++) {
            const int nb = wn * 32 + ni * 16;
            for (int e = 0; e < 16; e++)
                wfr[ni].h[e] = Wt[0][(kb + (e < 8 ? e : e + 8)) * WTSTR + nb + rsel];
        }
#pragma unroll
        for (int mi = 0; mi < 4; mi++)
#pragma unroll
            for (int ni = 0; ni < 2; ni++)
                acc[mi][ni] = __builtin_amdgcn_wmma_f32_16x16x32_bf16(
                    false, afr[mi].v, false, wfr[ni].v,
                    (short)0, acc[mi][ni], false, false);
        __syncthreads();
    }
#endif

    // ---- epilogue: C[m][n] = acc + bias[n] ----
    const int hi = lane >> 4;
#pragma unroll
    for (int ni = 0; ni < 2; ni++) {
        const int n  = n0 + wn * 32 + ni * 16 + rsel;
        const float bv = bias[n];
#pragma unroll
        for (int mi = 0; mi < 4; mi++) {
            const int mb = m0 + wm * 64 + mi * 16 + 8 * hi;
#pragma unroll
            for (int r = 0; r < 8; r++)
                C[(size_t)(mb + r) * N + n] = acc[mi][ni][r] + bv;
        }
    }
}

// ---------------------------------------------------------------------------
// causal depthwise conv (D_CONV=4) + SiLU ->  bf16 A for the GRU-input GEMM
// xz layout: (M, 2048), x_proj = cols [0,1024)
// ---------------------------------------------------------------------------
__global__ __launch_bounds__(256)
void k_conv_silu(const float* __restrict__ xz, const float* __restrict__ cw,
                 const float* __restrict__ cb, __bf16* __restrict__ out)
{
    const int idx = blockIdx.x * 256 + threadIdx.x;   // < 4*1024*1024
    const int c = idx & 1023;
    const int t = (idx >> 10) & 1023;
    const int b = idx >> 20;
    float y = cb[c];
#pragma unroll
    for (int k = 0; k < 4; k++) {
        const int tt = t + k - 3;
        if (tt >= 0)
            y += xz[(size_t)(b * TLEN + tt) * 2048 + c] * cw[c * 4 + k];
    }
    out[(size_t)idx] = f2bf(y * sigmoidf_(y));
}

// ---------------------------------------------------------------------------
// gating:  gated = gruout * silu(z)   (z = xz cols [1024,2048)) -> bf16 A
// ---------------------------------------------------------------------------
__global__ __launch_bounds__(256)
void k_gate(const float* __restrict__ gruout, const float* __restrict__ xz,
            __bf16* __restrict__ out)
{
    const int idx = blockIdx.x * 256 + threadIdx.x;   // < 4096*1024
    const int m = idx >> 10;
    const int c = idx & 1023;
    const float z = xz[(size_t)m * 2048 + 1024 + c];
    out[(size_t)idx] = f2bf(gruout[idx] * z * sigmoidf_(z));
}

// ---------------------------------------------------------------------------
// Persistent GRU scan. Grid = 64 WGs x 384 threads (12 waves).
// WG g owns columns j in [g*16, g*16+16) of all three gate thirds.
// whh slice is held as bf16 B-fragments in REGISTERS for all 1024 steps.
// wave w: third = w/4, K-quarter = w%4 (256 K = 8 chained WMMAs/step).
// h ping-pongs through global; one atomic-counter grid barrier per step.
// A tile = h padded to M=16 (rows 4..15 zero -> EXEC stays all ones).
// ---------------------------------------------------------------------------
#define HSTR 1032   // lds_h row stride (bf16)

__global__ __launch_bounds__(384)
void k_gru_scan(const float* __restrict__ xih, const float* __restrict__ whh,
                const float* __restrict__ bhh, float* __restrict__ hbuf,
                float* __restrict__ gruout, unsigned* __restrict__ counter,
                unsigned base)
{
    __shared__ __bf16 lds_h[4 * HSTR];          // h as bf16, rows = batch
    __shared__ float  lds_part[3 * 4 * 4 * 16]; // [third][quarter][b][j]
    __shared__ float  lds_bhh[48];

    const int tid  = threadIdx.x;
    const int lane = tid & 31;
    const int wid  = tid >> 5;
    const int third   = wid >> 2;       // 0..2
    const int quarter = wid & 3;        // 0..3
    const int kbase0  = quarter * 256;
    const int g  = blockIdx.x;
    const int j0 = g * 16;

    if (tid < 48)
        lds_bhh[tid] = bhh[(tid >> 4) * D_INNER + j0 + (tid & 15)];

    // --- preload whh B-fragments into registers (lane<->N, elem<->K) ---
    AF wfrag[8];
#pragma unroll
    for (int c = 0; c < 8; c++) {
        const int kg = kbase0 + c * 32 + (lane >> 4) * 8;
#pragma unroll
        for (int e = 0; e < 16; e++)
            wfrag[c].h[e] =
                f2bf(whh[(size_t)(third * D_INNER + j0 + (lane & 15)) * D_INNER + kg + (e < 8 ? e : e + 8)]);
    }

    const int  row   = lane & 15;
    const int  kb    = (lane >> 4) * 8;
    const bool rowok = row < BATCH;
    __syncthreads();

    for (int t = 0; t < TLEN; t++) {
        float* hin  = hbuf + (t & 1) * (BATCH * D_INNER);
        float* hout = hbuf + ((t + 1) & 1) * (BATCH * D_INNER);

        // stage h (agent-scope loads) into LDS as bf16
        for (int idx = tid; idx < BATCH * D_INNER; idx += 384) {
            float hv = __hip_atomic_load(&hin[idx], __ATOMIC_RELAXED,
                                         __HIP_MEMORY_SCOPE_AGENT);
            lds_h[(idx >> 10) * HSTR + (idx & 1023)] = f2bf(hv);
        }
        __syncthreads();

        // matvec slice: 16x16 tile, K = 256 per wave
        v8f acc = (v8f){0.f,0.f,0.f,0.f,0.f,0.f,0.f,0.f};
#pragma unroll
        for (int c = 0; c < 8; c++) {
            AF a;
            if (rowok) {
                a.q[0] = *(const Q128*)&lds_h[row * HSTR + kbase0 + c * 32 + kb];
                a.q[1] = *(const Q128*)&lds_h[row * HSTR + kbase0 + c * 32 + kb + 16];
            } else {
                Q128 z; z.u[0] = z.u[1] = z.u[2] = z.u[3] = 0u;
                a.q[0] = z; a.q[1] = z;
            }
            acc = __builtin_amdgcn_wmma_f32_16x16x32_bf16(
                false, a.v, false, wfrag[c].v, (short)0, acc, false, false);
        }

        // dump rows 0..3 (C layout: lane<->N, vgpr<->M) for reduction
        if (lane < 16) {
#pragma unroll
            for (int r = 0; r < BATCH; r++)
                lds_part[((third * 4 + quarter) * 4 + r) * 16 + lane] = acc[r];
        }
        __syncthreads();

        // gates: 64 threads = (b, j)
        if (tid < 64) {
            const int b = tid >> 4, j = tid & 15;
            float hg[3];
#pragma unroll
            for (int th = 0; th < 3; th++) {
                float s = lds_bhh[th * 16 + j];
#pragma unroll
                for (int q = 0; q < 4; q++)
                    s += lds_part[((th * 4 + q) * 4 + b) * 16 + j];
                hg[th] = s;
            }
            const int m   = b * TLEN + t;
            const int col = j0 + j;
            const float* xg = xih + (size_t)m * (3 * D_INNER);
            const float rr = sigmoidf_(xg[col]              + hg[0]);
            const float zz = sigmoidf_(xg[D_INNER + col]    + hg[1]);
            const float nn = tanhf   (xg[2 * D_INNER + col] + rr * hg[2]);
            const float hp = __hip_atomic_load(&hin[b * D_INNER + col],
                                               __ATOMIC_RELAXED,
                                               __HIP_MEMORY_SCOPE_AGENT);
            const float hn = (1.0f - zz) * nn + zz * hp;
            __hip_atomic_store(&hout[b * D_INNER + col], hn,
                               __ATOMIC_RELAXED, __HIP_MEMORY_SCOPE_AGENT);
            gruout[(size_t)m * D_INNER + col] = hn;
        }

        // grid barrier (all 64 WGs)
        __threadfence();
        __syncthreads();
        if (tid == 0) {
            __hip_atomic_fetch_add(counter, 1u, __ATOMIC_RELEASE,
                                   __HIP_MEMORY_SCOPE_AGENT);
            const unsigned target = base + (unsigned)(t + 1) * 64u;
            while (__hip_atomic_load(counter, __ATOMIC_ACQUIRE,
                                     __HIP_MEMORY_SCOPE_AGENT) < target)
                __builtin_amdgcn_s_sleep(1);
        }
        __syncthreads();
    }
}

// ---------------------------------------------------------------------------
// host orchestration
// ---------------------------------------------------------------------------
extern "C" void kernel_launch(void* const* d_in, const int* in_sizes, int n_in,
                              void* d_out, int out_size, void* d_ws, size_t ws_size,
                              hipStream_t stream)
{
    (void)in_sizes; (void)n_in; (void)out_size; (void)ws_size;

    const float* x        = (const float*)d_in[0];
    const float* norm_w   = (const float*)d_in[1];
    const float* norm_b   = (const float*)d_in[2];
    const float* in_w     = (const float*)d_in[3];
    const float* in_b     = (const float*)d_in[4];
    const float* conv_w   = (const float*)d_in[5];
    const float* conv_b   = (const float*)d_in[6];
    const float* gru_wih  = (const float*)d_in[7];
    const float* gru_whh  = (const float*)d_in[8];
    const float* gru_bih  = (const float*)d_in[9];
    const float* gru_bhh  = (const float*)d_in[10];
    const float* out_w    = (const float*)d_in[11];
    const float* out_b    = (const float*)d_in[12];
    const float* normf_w  = (const float*)d_in[13];
    const float* normf_b  = (const float*)d_in[14];

    char* ws = (char*)d_ws;
    size_t off = 0;
    float*  resid   = (float*)(ws + off); off += (size_t)MROWS * D_MODEL * 4;      // 8 MB
    float*  hiddenB = (float*)(ws + off); off += (size_t)MROWS * D_MODEL * 4;      // 8 MB
    float*  xz      = (float*)(ws + off); off += (size_t)MROWS * 2048 * 4;         // 32 MB
    float*  xih     = (float*)(ws + off); off += (size_t)MROWS * 3072 * 4;         // 48 MB
    float*  gruo    = (float*)(ws + off); off += (size_t)MROWS * D_INNER * 4;      // 16 MB
    __bf16* Abf     = (__bf16*)(ws + off); off += (size_t)MROWS * D_INNER * 2;     // 8 MB
    __bf16* Wbf     = (__bf16*)(ws + off); off += (size_t)3072 * 1024 * 2;         // 6 MB (W^T)
    float*  hbuf    = (float*)(ws + off); off += 2 * BATCH * D_INNER * 4;          // 32 KB
    unsigned* counter = (unsigned*)(ws + off);

    // zero barrier counter (64 floats covers the padded region)
    k_zero<<<1, 64, 0, stream>>>((float*)counter, 64);

    for (int i = 0; i < 6; i++) {
        // residual + LN -> bf16 A
        k_resid_ln<<<MROWS, 256, 0, stream>>>(
            (i == 0) ? x : hiddenB, resid, Abf, nullptr,
            norm_w + (size_t)i * D_MODEL, norm_b + (size_t)i * D_MODEL,
            (i == 0) ? 1 : 0, 1);

        // in_proj GEMM: (4096,512) x W^T(512,2048) -> xz
        k_cvt_t<<<1024, 256, 0, stream>>>(in_w + (size_t)i * 2048 * 512, Wbf, 2048, 512);
        k_gemm<<<dim3(2048 / BN, MROWS / BM), 256, 0, stream>>>(
            Abf, Wbf, in_b + (size_t)i * 2048, xz, 2048, 512);

        // conv + silu -> bf16 A (4096 x 1024)
        k_conv_silu<<<(MROWS * D_INNER) / 256, 256, 0, stream>>>(
            xz, conv_w + (size_t)i * D_INNER * 4, conv_b + (size_t)i * D_INNER, Abf);

        // GRU input GEMM: (4096,1024) x W^T(1024,3072) -> xih
        k_cvt_t<<<1024, 256, 0, stream>>>(gru_wih + (size_t)i * 3072 * 1024, Wbf, 3072, 1024);
        k_gemm<<<dim3(3072 / BN, MROWS / BM), 256, 0, stream>>>(
            Abf, Wbf, gru_bih + (size_t)i * 3072, xih, 3072, 1024);

        // GRU scan (h state zeroed per layer)
        k_zero<<<32, 256, 0, stream>>>(hbuf, 2 * BATCH * D_INNER);
        k_gru_scan<<<64, 384, 0, stream>>>(
            xih, gru_whh + (size_t)i * 3072 * 1024, gru_bhh + (size_t)i * 3072,
            hbuf, gruo, counter, (unsigned)(i * 65536));

        // gate with silu(z) -> bf16 A
        k_gate<<<(MROWS * D_INNER) / 256, 256, 0, stream>>>(gruo, xz, Abf);

        // out_proj GEMM: (4096,1024) x W^T(1024,512) -> hiddenB
        k_cvt_t<<<1024, 256, 0, stream>>>(out_w + (size_t)i * 512 * 1024, Wbf, 512, 1024);
        k_gemm<<<dim3(512 / BN, MROWS / BM), 256, 0, stream>>>(
            Abf, Wbf, out_b + (size_t)i * 512, hiddenB, 512, 1024);
    }

    // final residual + LN -> f32 output
    k_resid_ln<<<MROWS, 256, 0, stream>>>(
        hiddenB, resid, nullptr, (float*)d_out, normf_w, normf_b, 0, 0);
}
